// FlaxLlamaAttention_25580825215175
// MI455X (gfx1250) — matrix operand order repaired and checked by reading, not compiled, not run on hardware
//
#include <hip/hip_runtime.h>

#define S_LEN  2048
#define DMODEL 4096
#define NHEAD  32
#define HDIM   128

typedef __attribute__((ext_vector_type(4)))  unsigned int u32x4;
typedef __attribute__((ext_vector_type(8)))  float        v8f;
typedef __attribute__((ext_vector_type(16))) __bf16       v16bf;

static __device__ __forceinline__ unsigned short f2bf(float f) {
  unsigned u = __float_as_uint(f);
  unsigned r = ((u >> 16) & 1u) + 0x7FFFu;   // round-to-nearest-even
  return (unsigned short)((u + r) >> 16);
}
static __device__ __forceinline__ float bf2f(unsigned short h) {
  return __uint_as_float(((unsigned)h) << 16);
}
static __device__ __forceinline__ v16bf mkfrag(u32x4 lo, u32x4 hi) {
  union { u32x4 u[2]; v16bf v; } x;
  x.u[0] = lo; x.u[1] = hi;
  return x.v;
}

// ---- CDNA5 async global->LDS DMA (VGLOBAL op 98, tracked by ASYNCcnt) ----
static __device__ __forceinline__ void async_b128(unsigned lds, unsigned voff,
                                                  const void* sbase) {
  asm volatile("global_load_async_to_lds_b128 %0, %1, %2"
               :: "v"(lds), "v"(voff), "s"(sbase) : "memory");
}
static __device__ __forceinline__ void wait_async0() {
  asm volatile("s_wait_asynccnt 0" ::: "memory");
}

// DPP row-rotate (within 16-lane row) — pure-VALU cross-lane reductions
template <int CTRL>
static __device__ __forceinline__ float dppmov(float x) {
  return __int_as_float(
      __builtin_amdgcn_update_dpp(0, __float_as_int(x), CTRL, 0xf, 0xf, true));
}
static __device__ __forceinline__ float rowmax16(float x) {
  x = fmaxf(x, dppmov<0x121>(x));
  x = fmaxf(x, dppmov<0x122>(x));
  x = fmaxf(x, dppmov<0x124>(x));
  x = fmaxf(x, dppmov<0x128>(x));
  return x;
}
static __device__ __forceinline__ float rowsum16(float x) {
  x += dppmov<0x121>(x);
  x += dppmov<0x122>(x);
  x += dppmov<0x124>(x);
  x += dppmov<0x128>(x);
  return x;
}

// ---------------------------------------------------------------- x -> bf16
__global__ __launch_bounds__(256) void f32_to_bf16_kernel(
    const float* __restrict__ in, unsigned short* __restrict__ out, long n) {
  long i = ((long)blockIdx.x * blockDim.x + threadIdx.x) * 8;
  if (i >= n) return;
  float4 a = *reinterpret_cast<const float4*>(in + i);
  float4 b = *reinterpret_cast<const float4*>(in + i + 4);
  u32x4 p;
  p[0] = ((unsigned)f2bf(a.y) << 16) | f2bf(a.x);
  p[1] = ((unsigned)f2bf(a.w) << 16) | f2bf(a.z);
  p[2] = ((unsigned)f2bf(b.y) << 16) | f2bf(b.x);
  p[3] = ((unsigned)f2bf(b.w) << 16) | f2bf(b.z);
  *reinterpret_cast<u32x4*>(out + i) = p;
}

// ------------------------------------------- W [K,N] f32 -> WT [N,K] bf16
__global__ __launch_bounds__(256) void w_to_bf16t_kernel(
    const float* __restrict__ W, unsigned short* __restrict__ WT, int K, int N) {
  __shared__ unsigned short T[64 * 72];
  const int tid = threadIdx.x;
  const int k0  = blockIdx.y * 64;
  const int n0  = blockIdx.x * 64;
#pragma unroll
  for (int t = 0; t < 4; t++) {
    int idx = tid + t * 256;
    int row = idx >> 4;
    int c4  = (idx & 15) * 4;
    float4 f = *reinterpret_cast<const float4*>(&W[(size_t)(k0 + row) * N + n0 + c4]);
    T[(c4 + 0) * 72 + row] = f2bf(f.x);
    T[(c4 + 1) * 72 + row] = f2bf(f.y);
    T[(c4 + 2) * 72 + row] = f2bf(f.z);
    T[(c4 + 3) * 72 + row] = f2bf(f.w);
  }
  __syncthreads();
#pragma unroll
  for (int t = 0; t < 2; t++) {
    int idx = tid + t * 256;
    int row = idx >> 3;
    int col = (idx & 7) * 8;
    *reinterpret_cast<u32x4*>(&WT[(size_t)(n0 + row) * K + k0 + col]) =
        *reinterpret_cast<const u32x4*>(&T[row * 72 + col]);
  }
}

// ---------------------------------------------------------------- GEMM
// C[M,N] = A(bf16 [M,K]) * B(bf16 [N,K] = W^T). Block 128x128, BK=64,
// 8 waves (4x2), wave tile 32x64, async-DMA ping-pong LDS, 1 barrier/stage.
// MODE 0: bf16 [M,N]; MODE 1: bf16 [N,M] (V^T); MODE 2: f32 [M,N]
template <int MODE>
__global__ __launch_bounds__(256) void gemm_bf16t_kernel(
    const unsigned short* __restrict__ A, const unsigned short* __restrict__ B,
    void* __restrict__ out, int M, int N, int K) {
  __shared__ unsigned short As[2][128 * 64];  // [m][k]
  __shared__ unsigned short Bs[2][128 * 64];  // [n][k]

  const int tid  = threadIdx.x;
  const int lane = tid & 31;
  const int wave = tid >> 5;
  const int wm   = wave >> 1;
  const int wn   = wave & 1;
  const int m0   = blockIdx.y * 128;
  const int n0   = blockIdx.x * 128;

  v8f acc[2][4];
#pragma unroll
  for (int a = 0; a < 2; a++)
#pragma unroll
    for (int b = 0; b < 4; b++)
#pragma unroll
      for (int r = 0; r < 8; r++) acc[a][b][r] = 0.0f;

  const int am  = lane & 15;
  const int aks = (lane >> 4) * 8;
  const int bn  = lane & 15;
  const int bks = (lane >> 4) * 8;

  const int grow[4] = {(tid + 0) >> 3, (tid + 256) >> 3, (tid + 512) >> 3, (tid + 768) >> 3};
  const int gcol    = (tid & 7) * 8;

  auto stage = [&](int buf, int kn) {
#pragma unroll
    for (int t = 0; t < 4; t++) {
      unsigned lA = (unsigned)(uintptr_t)&As[buf][grow[t] * 64 + gcol];
      unsigned oA = (unsigned)(((size_t)(m0 + grow[t]) * K + kn + gcol) * 2);
      async_b128(lA, oA, A);
      unsigned lB = (unsigned)(uintptr_t)&Bs[buf][grow[t] * 64 + gcol];
      unsigned oB = (unsigned)(((size_t)(n0 + grow[t]) * K + kn + gcol) * 2);
      async_b128(lB, oB, B);
    }
  };

  stage(0, 0);
  wait_async0();
  __syncthreads();

  const int nK = K / 64;
  for (int i = 0; i < nK; i++) {
    const int cur = i & 1;
    const bool more = (i + 1) < nK;
    if (more) stage(cur ^ 1, (i + 1) * 64);

#pragma unroll
    for (int ks = 0; ks < 2; ks++) {
      const int kk = ks * 32;
      v16bf afrag[2], bfrag[4];
#pragma unroll
      for (int mt = 0; mt < 2; mt++) {
        int row = wm * 32 + mt * 16 + am;
        u32x4 lo = *reinterpret_cast<const u32x4*>(&As[cur][row * 64 + kk + aks]);
        u32x4 hi = *reinterpret_cast<const u32x4*>(&As[cur][row * 64 + kk + aks + 16]);
        afrag[mt] = mkfrag(lo, hi);
      }
#pragma unroll
      for (int nt = 0; nt < 4; nt++) {
        int rown = wn * 64 + nt * 16 + bn;
        u32x4 lo = *reinterpret_cast<const u32x4*>(&Bs[cur][rown * 64 + kk + bks]);
        u32x4 hi = *reinterpret_cast<const u32x4*>(&Bs[cur][rown * 64 + kk + bks + 16]);
        bfrag[nt] = mkfrag(lo, hi);
      }
#pragma unroll
      for (int mt = 0; mt < 2; mt++)
#pragma unroll
        for (int nt = 0; nt < 4; nt++)
          acc[mt][nt] = __builtin_amdgcn_wmma_f32_16x16x32_bf16(
              false, afrag[mt], false, bfrag[nt], (short)0, acc[mt][nt], false, false);
    }
    if (more) wait_async0();
    __syncthreads();
  }

  const int cn   = lane & 15;
  const int rsel = (lane >> 4) * 8;
#pragma unroll
  for (int mt = 0; mt < 2; mt++)
#pragma unroll
    for (int nt = 0; nt < 4; nt++)
#pragma unroll
      for (int r = 0; r < 8; r++) {
        int row = m0 + wm * 32 + mt * 16 + rsel + r;
        int col = n0 + wn * 64 + nt * 16 + cn;
        float v = acc[mt][nt][r];
        if (MODE == 0) {
          ((unsigned short*)out)[(size_t)row * N + col] = f2bf(v);
        } else if (MODE == 1) {
          ((unsigned short*)out)[(size_t)col * M + row] = f2bf(v);
        } else {
          ((float*)out)[(size_t)row * N + col] = v;
        }
      }
}

// -------------------------------- RoPE (in-place; Q pre-scaled by 1/sqrt(hd))
__global__ __launch_bounds__(256) void rope_kernel(unsigned short* __restrict__ Q,
                                                   unsigned short* __restrict__ Kk) {
  long idx = (long)blockIdx.x * blockDim.x + threadIdx.x;
  if (idx >= (long)S_LEN * DMODEL / 2) return;
  int pair = (int)(idx % (DMODEL / 2));  // h*64 + j
  int s    = (int)(idx / (DMODEL / 2));
  int j    = pair & 63;
  float inv = __powf(10000.0f, -(float)(2 * j) * (1.0f / 128.0f));
  float ang = (float)s * inv;
  float sn, cs;
  sincosf(ang, &sn, &cs);
  size_t base = (size_t)s * DMODEL + 2 * pair;
  const float qscale = 0.08838834764831845f;  // 1/sqrt(128)

  float a = bf2f(Q[base]), b = bf2f(Q[base + 1]);
  Q[base]     = f2bf((a * cs - b * sn) * qscale);
  Q[base + 1] = f2bf((a * sn + b * cs) * qscale);

  a = bf2f(Kk[base]); b = bf2f(Kk[base + 1]);
  Kk[base]     = f2bf(a * cs - b * sn);
  Kk[base + 1] = f2bf(a * sn + b * cs);
}

// ---------------------------------------------------------------- flash attention, BK=64
__global__ __launch_bounds__(256) void attn_kernel(
    const unsigned short* __restrict__ Q, const unsigned short* __restrict__ Kg,
    const unsigned short* __restrict__ VT, unsigned short* __restrict__ O) {
  __shared__ unsigned short Ks[2][64 * 128];   // [key][d]
  __shared__ unsigned short Vs[2][128 * 64];   // [d][key]
  __shared__ unsigned short Ps[8][16 * 64];    // per-wave P tile

  const int tid  = threadIdx.x;
  const int lane = tid & 31;
  const int wave = tid >> 5;
  const int h    = blockIdx.y;
  const int q0   = blockIdx.x * 128;
  const int qr0  = q0 + wave * 16;

  const int am  = lane & 15;
  const int aks = (lane >> 4) * 8;
  const int bn  = lane & 15;
  const int bks = (lane >> 4) * 8;
  const int hi8 = (lane >> 4) * 8;
  const int cl  = lane & 15;

  const int krow[4] = {(tid + 0) >> 4, (tid + 256) >> 4, (tid + 512) >> 4, (tid + 768) >> 4};
  const int kcol    = (tid & 15) * 8;
  const int vrow[4] = {(tid + 0) >> 3, (tid + 256) >> 3, (tid + 512) >> 3, (tid + 768) >> 3};
  const int vcol    = (tid & 7) * 8;

  auto stageKV = [&](int buf, int kp) {
#pragma unroll
    for (int t = 0; t < 4; t++) {
      unsigned lK = (unsigned)(uintptr_t)&Ks[buf][krow[t] * 128 + kcol];
      unsigned oK = (unsigned)(((size_t)(kp + krow[t]) * DMODEL + h * HDIM + kcol) * 2);
      async_b128(lK, oK, Kg);
      unsigned lV = (unsigned)(uintptr_t)&Vs[buf][vrow[t] * 64 + vcol];
      unsigned oV = (unsigned)(((size_t)(h * HDIM + vrow[t]) * S_LEN + kp + vcol) * 2);
      async_b128(lV, oV, VT);
    }
  };

  // Q fragments (pre-scaled by 1/sqrt(hd) in rope)
  v16bf qf[4];
  const unsigned short* qbase = &Q[(size_t)(qr0 + am) * DMODEL + h * HDIM];
#pragma unroll
  for (int kc = 0; kc < 4; kc++) {
    u32x4 lo = *reinterpret_cast<const u32x4*>(qbase + kc * 32 + aks);
    u32x4 hi = *reinterpret_cast<const u32x4*>(qbase + kc * 32 + aks + 16);
    qf[kc] = mkfrag(lo, hi);
  }

  v8f o[8];
#pragma unroll
  for (int d = 0; d < 8; d++)
#pragma unroll
    for (int r = 0; r < 8; r++) o[d][r] = 0.0f;
  float mrow[8], lrow[8];
#pragma unroll
  for (int r = 0; r < 8; r++) { mrow[r] = -1e30f; lrow[r] = 0.0f; }

  const int nkb = (q0 + 128) / 64;

  stageKV(0, 0);
  wait_async0();
  __syncthreads();

#define SOFTMAX_BLOCK(MASKED)                                                 \
  _Pragma("unroll")                                                           \
  for (int r = 0; r < 8; r++) {                                               \
    float s0 = sc[0][r], s1 = sc[1][r], s2 = sc[2][r], s3 = sc[3][r];         \
    if (MASKED) {                                                             \
      int qrow = qr0 + hi8 + r;                                               \
      if (kp0 + cl > qrow)      s0 = -1e30f;                                  \
      if (kp0 + cl + 16 > qrow) s1 = -1e30f;                                  \
      if (kp0 + cl + 32 > qrow) s2 = -1e30f;                                  \
      if (kp0 + cl + 48 > qrow) s3 = -1e30f;                                  \
    }                                                                         \
    float rm   = rowmax16(fmaxf(fmaxf(s0, s1), fmaxf(s2, s3)));               \
    float mnew = fmaxf(mrow[r], rm);                                          \
    float alpha = __expf(mrow[r] - mnew);                                     \
    float p0 = __expf(s0 - mnew), p1 = __expf(s1 - mnew);                     \
    float p2 = __expf(s2 - mnew), p3 = __expf(s3 - mnew);                     \
    float rs = rowsum16((p0 + p1) + (p2 + p3));                               \
    lrow[r] = lrow[r] * alpha + rs;                                           \
    mrow[r] = mnew;                                                           \
    _Pragma("unroll")                                                         \
    for (int dt = 0; dt < 8; dt++) o[dt][r] *= alpha;                         \
    Ps[wave][(hi8 + r) * 64 + cl]      = f2bf(p0);                            \
    Ps[wave][(hi8 + r) * 64 + cl + 16] = f2bf(p1);                            \
    Ps[wave][(hi8 + r) * 64 + cl + 32] = f2bf(p2);                            \
    Ps[wave][(hi8 + r) * 64 + cl + 48] = f2bf(p3);                            \
  }

  for (int kb = 0; kb < nkb; kb++) {
    const int cur = kb & 1;
    const int kp0 = kb * 64;
    const bool more = (kb + 1) < nkb;
    if (more) stageKV(cur ^ 1, kp0 + 64);

    if (kp0 <= qr0 + 15) {  // wave-uniform causal skip
      v8f sc[4];
#pragma unroll
      for (int nt = 0; nt < 4; nt++)
#pragma unroll
        for (int r = 0; r < 8; r++) sc[nt][r] = 0.0f;
      // kc-outer / nt-inner: batch 4 independent B-frag loads per k-step,
      // then 4 independent WMMAs (distinct accumulators) — same shape that
      // schedules well in the GEMM.
#pragma unroll
      for (int kc = 0; kc < 4; kc++) {
        v16bf bfrag[4];
#pragma unroll
        for (int nt = 0; nt < 4; nt++) {
          u32x4 lo = *reinterpret_cast<const u32x4*>(
              &Ks[cur][(nt * 16 + bn) * 128 + kc * 32 + bks]);
          u32x4 hi = *reinterpret_cast<const u32x4*>(
              &Ks[cur][(nt * 16 + bn) * 128 + kc * 32 + bks + 16]);
          bfrag[nt] = mkfrag(lo, hi);
        }
#pragma unroll
        for (int nt = 0; nt < 4; nt++)
          sc[nt] = __builtin_amdgcn_wmma_f32_16x16x32_bf16(
              false, qf[kc], false, bfrag[nt], (short)0, sc[nt], false, false);
      }
      if (kp0 + 63 <= qr0) {
        SOFTMAX_BLOCK(false)
      } else {
        SOFTMAX_BLOCK(true)
      }
      // same-wave LDS round-trip (per-wave DS ops are in-order; compiler emits waits)
      v16bf pf[2];
#pragma unroll
      for (int pc = 0; pc < 2; pc++) {
        u32x4 plo = *reinterpret_cast<const u32x4*>(&Ps[wave][am * 64 + pc * 32 + aks]);
        u32x4 phi = *reinterpret_cast<const u32x4*>(&Ps[wave][am * 64 + pc * 32 + aks + 16]);
        pf[pc] = mkfrag(plo, phi);
      }
#pragma unroll
      for (int dt = 0; dt < 8; dt++) {
#pragma unroll
        for (int pc = 0; pc < 2; pc++) {
          u32x4 lo = *reinterpret_cast<const u32x4*>(
              &Vs[cur][(dt * 16 + bn) * 64 + pc * 32 + bks]);
          u32x4 hi = *reinterpret_cast<const u32x4*>(
              &Vs[cur][(dt * 16 + bn) * 64 + pc * 32 + bks + 16]);
          o[dt] = __builtin_amdgcn_wmma_f32_16x16x32_bf16(
              false, pf[pc], false, mkfrag(lo, hi), (short)0, o[dt], false, false);
        }
      }
    }

    if (more) wait_async0();
    __syncthreads();
  }
#undef SOFTMAX_BLOCK

  // normalize + store bf16
#pragma unroll
  for (int r = 0; r < 8; r++) {
    float invl = 1.0f / lrow[r];
    int row = qr0 + hi8 + r;
#pragma unroll
    for (int dt = 0; dt < 8; dt++) {
      int col = h * HDIM + dt * 16 + cl;
      O[(size_t)row * DMODEL + col] = f2bf(o[dt][r] * invl);
    }
  }
}

// ---------------------------------------------------------------- launch
extern "C" void kernel_launch(void* const* d_in, const int* in_sizes, int n_in,
                              void* d_out, int out_size, void* d_ws, size_t ws_size,
                              hipStream_t stream) {
  (void)in_sizes; (void)n_in; (void)out_size; (void)ws_size;
  const float* x  = (const float*)d_in[0];
  const float* wq = (const float*)d_in[1];
  const float* wk = (const float*)d_in[2];
  const float* wv = (const float*)d_in[3];
  const float* wo = (const float*)d_in[4];

  const long nx = (long)S_LEN * DMODEL;
  unsigned short* xb = (unsigned short*)d_ws;    // x  bf16        [S,D]
  unsigned short* qb = xb + nx;                  // Q  bf16        [S,D]
  unsigned short* kb = qb + nx;                  // K  bf16        [S,D]
  unsigned short* vt = kb + nx;                  // V^T bf16       [D,S]
  unsigned short* ab = vt + nx;                  // attn out bf16  [S,D]
  unsigned short* wt = ab + nx;                  // W^T bf16       [N,K] (reused)

  f32_to_bf16_kernel<<<(unsigned)((nx / 8 + 255) / 256), 256, 0, stream>>>(x, xb, nx);

  dim3 gw(DMODEL / 64, DMODEL / 64);
  dim3 gg(DMODEL / 128, S_LEN / 128);

  w_to_bf16t_kernel<<<gw, 256, 0, stream>>>(wq, wt, DMODEL, DMODEL);
  gemm_bf16t_kernel<0><<<gg, 256, 0, stream>>>(xb, wt, qb, S_LEN, DMODEL, DMODEL);
  w_to_bf16t_kernel<<<gw, 256, 0, stream>>>(wk, wt, DMODEL, DMODEL);
  gemm_bf16t_kernel<0><<<gg, 256, 0, stream>>>(xb, wt, kb, S_LEN, DMODEL, DMODEL);
  w_to_bf16t_kernel<<<gw, 256, 0, stream>>>(wv, wt, DMODEL, DMODEL);
  gemm_bf16t_kernel<1><<<gg, 256, 0, stream>>>(xb, wt, vt, S_LEN, DMODEL, DMODEL);

  const long npairs = (long)S_LEN * DMODEL / 2;
  rope_kernel<<<(unsigned)((npairs + 255) / 256), 256, 0, stream>>>(qb, kb);

  dim3 ga(S_LEN / 128, NHEAD);
  attn_kernel<<<ga, 256, 0, stream>>>(qb, kb, vt, ab);

  w_to_bf16t_kernel<<<gw, 256, 0, stream>>>(wo, wt, DMODEL, DMODEL);
  gemm_bf16t_kernel<2><<<gg, 256, 0, stream>>>(ab, wt, d_out, S_LEN, DMODEL, DMODEL);
}
